// HierarchicalSparseAttentionRef_54829552500810
// MI455X (gfx1250) — compile-verified, matrix operand order checked by reading, not run on hardware
//
#include <hip/hip_runtime.h>
#include <hip/hip_bf16.h>
#include <cstdint>

typedef __attribute__((ext_vector_type(16))) _Float16 v16h;
typedef __attribute__((ext_vector_type(2)))  __fp16   v2fp16;
typedef __attribute__((ext_vector_type(8)))  float    v8f;

#define D_MODEL 1024
#define N_SEQ   2048
#define XY_LEN  4095      // 2N-1 rows of concat(query, y) per batch
#define N_HEADS 16
#define D_HEAD  64
#define RADIUS_ 16
#define N_SWA   33
#define N_KEYS  44        // 33 SWA + 11 hierarchical levels

// GEMM tiling: block computes 64(M) x 128(N), K-steps of 32, double-buffered LDS.
#define BM 64
#define BN 128
#define BK 32
#define SAS 36            // LDS row stride in halves: 72 B/row -> 8B-aligned dword chunks

// Single-instruction f32 pair -> packed f16 dword (v_cvt_pk_rtz_f16_f32).
__device__ __forceinline__ uint32_t pack2(float a, float b) {
  union { v2fp16 v; uint32_t u; } x;
  x.v = __builtin_amdgcn_cvt_pkrtz(a, b);
  return x.u;
}

// K index inside a 16x32 f16 A-fragment for dword r (0..7), laneHi = lane>>4.
// ISA 7.12.2: VGPR0..3 hold K = 8*laneHi + {0..7}; VGPR4..7 hold K = 16 + 8*laneHi + {0..7}.
__device__ __forceinline__ int fragK(int r, int laneHi) {
  return (r < 4) ? (laneHi * 8 + 2 * r) : (16 + laneHi * 8 + 2 * (r - 4));
}

// C = X @ W^T (+bias). X rows optionally gathered from concat(query, y).
template<bool XHALF, bool GATHER, bool BIAS, bool OUTF32>
__global__ __launch_bounds__(256)
void gemm_xwt(const void* __restrict__ Xa_, const float* __restrict__ Xb,
              const float* __restrict__ W, const float* __restrict__ bias,
              void* __restrict__ Out, int M_total)
{
  __shared__ __align__(16) _Float16 sA[2][BM * SAS];
  __shared__ __align__(16) _Float16 sB[2][BN * SAS];

  const int tid    = threadIdx.x;
  const int lane   = tid & 31;
  const int laneHi = lane >> 4;
  const int laneLo = lane & 15;
  const int wave   = tid >> 5;      // 8 waves
  const int wm     = wave >> 2;     // 0..1  -> 32-row strip
  const int wn     = wave & 3;      // 0..3  -> 32-col strip
  const int mBase  = blockIdx.y * BM;
  const int nBase  = blockIdx.x * BN;

  // ---- k-invariant staging addresses (hoisted: no division/branching in loop) ----
  // X tile: 64 rows x 8 four-elem chunks = 512 chunks; 2 per thread.
  int xrow[2], xc4[2];
  const float*    xpf[2];
  const _Float16* xph[2];
#pragma unroll
  for (int i = 0; i < 2; ++i) {
    int e = tid + i * 256;
    xrow[i] = e >> 3; xc4[i] = e & 7;
    int gm = mBase + xrow[i];
    if (gm > M_total - 1) gm = M_total - 1;        // clamp: branch-free OOB loads
    if constexpr (GATHER) {
      int bb = gm / XY_LEN, r = gm - bb * XY_LEN;
      xpf[i] = (r < N_SEQ) ? ((const float*)Xa_ + ((size_t)bb * N_SEQ + r) * D_MODEL)
                           : (Xb + ((size_t)bb * (N_SEQ - 1) + (r - N_SEQ)) * D_MODEL);
    } else if constexpr (XHALF) {
      xph[i] = (const _Float16*)Xa_ + (size_t)gm * D_MODEL;
    } else {
      xpf[i] = (const float*)Xa_ + (size_t)gm * D_MODEL;
    }
  }
  // W tile: 128 rows x 8 chunks = 1024 chunks; 4 per thread.
  int wrow[4], wc4[4];
  const float* wp[4];
#pragma unroll
  for (int i = 0; i < 4; ++i) {
    int e = tid + i * 256;
    wrow[i] = e >> 3; wc4[i] = e & 7;
    wp[i] = W + (size_t)(nBase + wrow[i]) * D_MODEL;
  }

  float4 xf[2]; uint2 xh[2]; float4 wf[4];
  auto loadTiles = [&](int kk) {                    // global -> regs (b128 loads)
#pragma unroll
    for (int i = 0; i < 2; ++i) {
      if constexpr (XHALF) xh[i] = *(const uint2*)(xph[i] + kk + xc4[i] * 4);
      else                 xf[i] = *(const float4*)(xpf[i] + kk + xc4[i] * 4);
    }
#pragma unroll
    for (int i = 0; i < 4; ++i) wf[i] = *(const float4*)(wp[i] + kk + wc4[i] * 4);
  };
  auto storeTiles = [&](int p) {                    // regs -> LDS (f16, b64 stores)
#pragma unroll
    for (int i = 0; i < 2; ++i) {
      uint2 d;
      if constexpr (XHALF) d = xh[i];
      else { d.x = pack2(xf[i].x, xf[i].y); d.y = pack2(xf[i].z, xf[i].w); }
      *(uint2*)&sA[p][xrow[i] * SAS + xc4[i] * 4] = d;
    }
#pragma unroll
    for (int i = 0; i < 4; ++i) {
      uint2 d; d.x = pack2(wf[i].x, wf[i].y); d.y = pack2(wf[i].z, wf[i].w);
      *(uint2*)&sB[p][wrow[i] * SAS + wc4[i] * 4] = d;
    }
  };

  v8f acc[2][2] = {};
  loadTiles(0);
  int buf = 0;
  for (int kk = 0; kk < D_MODEL; kk += BK) {
    storeTiles(buf);
    __syncthreads();
    if (kk + BK < D_MODEL) loadTiles(kk + BK);      // next tile in flight over compute

    // ---- fragments per ISA VGPR layout, then 4 WMMAs ----
    union AFrag { v16h v; uint32_t u[8]; } a[2], bfr[2];
#pragma unroll
    for (int mt = 0; mt < 2; ++mt) {
      int rowA = wm * 32 + mt * 16 + laneLo;
#pragma unroll
      for (int r = 0; r < 8; ++r)
        a[mt].u[r] = *(const uint32_t*)&sA[buf][rowA * SAS + fragK(r, laneHi)];
    }
#pragma unroll
    for (int nt = 0; nt < 2; ++nt) {
      int rowB = wn * 32 + nt * 16 + laneLo;        // B[k][n] = W[n][k]: same interleave
#pragma unroll
      for (int r = 0; r < 8; ++r)
        bfr[nt].u[r] = *(const uint32_t*)&sB[buf][rowB * SAS + fragK(r, laneHi)];
    }
#pragma unroll
    for (int mt = 0; mt < 2; ++mt)
#pragma unroll
      for (int nt = 0; nt < 2; ++nt)
        acc[mt][nt] = __builtin_amdgcn_wmma_f32_16x16x32_f16(
            false, a[mt].v, false, bfr[nt].v, (short)0, acc[mt][nt], false, false);
    buf ^= 1;
  }

  // ---- store: C/D layout M = r + 8*laneHi, N = laneLo ----
#pragma unroll
  for (int mt = 0; mt < 2; ++mt) {
#pragma unroll
    for (int nt = 0; nt < 2; ++nt) {
#pragma unroll
      for (int r = 0; r < 8; ++r) {
        int gm = mBase + wm * 32 + mt * 16 + r + 8 * laneHi;
        int gn = nBase + wn * 32 + nt * 16 + laneLo;
        if (gm < M_total) {
          float v = acc[mt][nt][r];
          if (BIAS) v += bias[gn];
          if (OUTF32)
            ((float*)Out)[(size_t)gm * D_MODEL + gn] = v;
          else
            ((_Float16*)Out)[(size_t)gm * D_MODEL + gn] = (_Float16)v;
        }
      }
    }
  }
}

// Sparse attention: one 64-thread block per (b, h, n). 44 keys/query.
__global__ __launch_bounds__(64)
void attn_sparse(const _Float16* __restrict__ Qp, const _Float16* __restrict__ Kp,
                 const _Float16* __restrict__ Vp, _Float16* __restrict__ Op)
{
  const int gb = blockIdx.x;                 // ((b*H)+h)*N + n
  const int n  = gb & (N_SEQ - 1);
  const int h  = (gb >> 11) & (N_HEADS - 1);
  const int b  = gb >> 15;
  const int t  = threadIdx.x;                // 0..63

  __shared__ float lq[D_HEAD];
  __shared__ float lg[N_KEYS];
  __shared__ int   kidx[N_KEYS];

  lq[t] = (float)Qp[((size_t)(b * N_SEQ + n)) * D_MODEL + h * D_HEAD + t];

  if (t < N_KEYS) {
    int idx;
    if (t < N_SWA) {                         // sliding window, clipped index
      int j = n - RADIUS_ + t;
      idx = j < 0 ? 0 : (j >= N_SEQ ? N_SEQ - 1 : j);
    } else {                                 // hierarchical tree level (t-33)
      int lvl = t - N_SWA;
      int cur = n, res = -1;
      for (int l = 0; l <= lvl; ++l) {
        int nxt = (l == 0) ? (cur ^ 1) : (((cur >> 1) + N_SEQ) ^ 1);
        if (nxt > 2 * N_SEQ - 3) { res = -1; break; }
        res = nxt; cur = nxt;
      }
      // reference leaves -1 which wraps to xy row 4094 (unmasked)
      idx = (res < 0) ? (2 * N_SEQ - 2) : res;
    }
    kidx[t] = idx;
  }
  __syncthreads();

  if (t < N_KEYS) {
    const uint32_t* k2 = (const uint32_t*)(Kp + ((size_t)b * XY_LEN + kidx[t]) * D_MODEL + h * D_HEAD);
    float dot = 0.f;
#pragma unroll
    for (int i = 0; i < D_HEAD / 2; ++i) {   // dword half-pairs -> vectorizable loads
      union { uint32_t u; _Float16 hh[2]; } c; c.u = k2[i];
      dot += lq[2 * i] * (float)c.hh[0] + lq[2 * i + 1] * (float)c.hh[1];
    }
    bool masked = false;
    if (t < N_SWA) { int j = n - RADIUS_ + t; masked = (j < 0) || (j >= N_SEQ); }
    lg[t] = masked ? -3.0e38f : dot * 0.125f;   // scale = 1/sqrt(64)
  }
  __syncthreads();

  float mx = -3.0e38f;
  for (int k = 0; k < N_KEYS; ++k) mx = fmaxf(mx, lg[k]);
  __syncthreads();
  if (t < N_KEYS) lg[t] = __expf(lg[t] - mx);
  __syncthreads();

  float sum = 0.f;
  for (int k = 0; k < N_KEYS; ++k) sum += lg[k];
  float acc = 0.f;
  for (int k = 0; k < N_KEYS; ++k) {
    const _Float16* vrow = Vp + ((size_t)b * XY_LEN + kidx[k]) * D_MODEL + h * D_HEAD;
    acc += lg[k] * (float)vrow[t];             // coalesced over t
  }
  Op[((size_t)(b * N_SEQ + n)) * D_MODEL + h * D_HEAD + t] = (_Float16)(acc / sum);
}

extern "C" void kernel_launch(void* const* d_in, const int* in_sizes, int n_in,
                              void* d_out, int out_size, void* d_ws, size_t ws_size,
                              hipStream_t stream) {
  (void)in_sizes; (void)n_in; (void)out_size; (void)ws_size;
  const float* query = (const float*)d_in[0];
  // d_in[1] (key) and d_in[2] (value) are unused by the reference.
  const float* y     = (const float*)d_in[3];
  const float* Wq    = (const float*)d_in[4];
  const float* Wk    = (const float*)d_in[5];
  const float* Wv    = (const float*)d_in[6];
  const float* Wo_w  = (const float*)d_in[7];
  const float* Wo_b  = (const float*)d_in[8];

  char* ws = (char*)d_ws;                                   // 48 MiB used
  _Float16* Qp   = (_Float16*)(ws);                         //  8 MiB: 4096 x 1024 f16
  _Float16* Kp   = (_Float16*)(ws + (size_t)8  * 1048576);  // 16 MiB: 8190(+2 pad) x 1024
  _Float16* Vp   = (_Float16*)(ws + (size_t)24 * 1048576);  // 16 MiB
  _Float16* Attn = (_Float16*)(ws + (size_t)40 * 1048576);  //  8 MiB

  dim3 blk(256);
  // Q = query @ Wq^T                       (M = B*N = 4096)
  gemm_xwt<false,false,false,false><<<dim3(8, 64),  blk, 0, stream>>>(query, nullptr, Wq, nullptr, Qp, 2 * N_SEQ);
  // K = concat(query,y) @ Wk^T             (M = B*(2N-1) = 8190)
  gemm_xwt<false,true ,false,false><<<dim3(8, 128), blk, 0, stream>>>(query, y, Wk, nullptr, Kp, 2 * XY_LEN);
  // V = concat(query,y) @ Wv^T
  gemm_xwt<false,true ,false,false><<<dim3(8, 128), blk, 0, stream>>>(query, y, Wv, nullptr, Vp, 2 * XY_LEN);
  // sparse attention (44 keys/query)
  attn_sparse<<<dim3(2 * N_HEADS * N_SEQ), dim3(64), 0, stream>>>(Qp, Kp, Vp, Attn);
  // out = attn @ Wo^T + b -> f32
  gemm_xwt<true ,false,true ,true ><<<dim3(8, 64),  blk, 0, stream>>>(Attn, nullptr, Wo_w, Wo_b, d_out, 2 * N_SEQ);
}
